// MultiHeadAttentionBlock_33595234189668
// MI455X (gfx1250) — compile-verified, matrix operand order checked by reading, not run on hardware
//
#include <hip/hip_runtime.h>

// ---------------------------------------------------------------------------
// MultiHeadAttentionBlock for MI455X (gfx1250, wave32, WMMA)
//   B=2, S=2048, D=1024, H=16, dk=64, causal, fp32 in/out.
// Pipeline: [QKV proj GEMM (wmma f16)] -> [flash attention (wmma f16)]
//           -> [output proj GEMM (wmma f16, f32 out)]
// ---------------------------------------------------------------------------

typedef __attribute__((ext_vector_type(16))) _Float16 v16h;
typedef __attribute__((ext_vector_type(8)))  _Float16 h8;
typedef __attribute__((ext_vector_type(8)))  float    v8f;
typedef __attribute__((ext_vector_type(4)))  int      v4i;

#define DMODEL 1024
#define SEQ    2048
#define NH     16
#define DKH    64
#define MROWS  4096   // B*S

// ---- optional CDNA5 async global->LDS staging (ASYNCcnt path) -------------
#if __has_builtin(__builtin_amdgcn_global_load_async_to_lds_b128)
#define USE_ASYNC_LDS 1
typedef __attribute__((address_space(1))) v4i* as1_v4i_p;
typedef __attribute__((address_space(3))) v4i* as3_v4i_p;
__device__ __forceinline__ void async_b128(const void* g, void* l) {
  __builtin_amdgcn_global_load_async_to_lds_b128(
      (as1_v4i_p)(size_t)g,
      (as3_v4i_p)(unsigned)(size_t)l,
      0, 0);
}
__device__ __forceinline__ void wait_async0() {
#if __has_builtin(__builtin_amdgcn_s_wait_asynccnt)
  __builtin_amdgcn_s_wait_asynccnt(0);
#else
  asm volatile("s_wait_asynccnt 0" ::: "memory");
#endif
}
#endif

__device__ __forceinline__ v8f wmma16(v16h a, v16h b, v8f c) {
  return __builtin_amdgcn_wmma_f32_16x16x32_f16(
      /*neg_a=*/false, a, /*neg_b=*/false, b,
      /*c_mod=*/(short)0, c, /*reuse_a=*/false, /*reuse_b=*/false);
}

// 16x32 f16 fragment in the ISA A/B register layout from a row-major tile.
// Lanes 0-15: row r16, K {0..7,16..23}; lanes 16-31: row r16, K {8..15,24..31}.
__device__ __forceinline__ v16h load_frag(const _Float16* base, int ld,
                                          int row0, int k0) {
  const int lane = threadIdx.x & 31;
  const int r16  = lane & 15;
  const int hl   = lane >> 4;
  const _Float16* p = base + (size_t)(row0 + r16) * ld + k0 + hl * 8;
  union { v16h v; h8 h[2]; } u;
  u.h[0] = *(const h8*)(p);
  u.h[1] = *(const h8*)(p + 16);
  return u.v;
}

__device__ __forceinline__ void cvt8(_Float16* dst, float4 a, float4 b) {
  union { h8 v; _Float16 e[8]; } u;
  u.e[0] = (_Float16)a.x; u.e[1] = (_Float16)a.y;
  u.e[2] = (_Float16)a.z; u.e[3] = (_Float16)a.w;
  u.e[4] = (_Float16)b.x; u.e[5] = (_Float16)b.y;
  u.e[6] = (_Float16)b.z; u.e[7] = (_Float16)b.w;
  *(h8*)dst = u.v;
}

// ---------------------------------------------------------------------------
// GEMM: Out = X(4096xD) * W(DxD)^T + bias
//   MODE 0: f16 out, [b][h][s][dk]   (Q, K)
//   MODE 1: f16 out, [b][h][dk][s]   (V transposed for the PV wmma)
//   MODE 2: f32 out, row-major       (final projection -> d_out)
// Block tile 128(M) x 256(N), K-step 32; 8 waves of 64x64 (16 wmma / K-step).
// ---------------------------------------------------------------------------
template <int MODE, bool INF16>
__global__ __launch_bounds__(256) void gemm_wmma(const void* __restrict__ Xv,
                                                 const float* __restrict__ W,
                                                 const float* __restrict__ bias,
                                                 void* __restrict__ Out) {
  constexpr int BK  = 32;
  constexpr int LDT = BK + 8;          // padded LDS stride (halves)
  __shared__ _Float16 Xs[128 * LDT];
  __shared__ _Float16 Ws[256 * LDT];

  const int tid  = threadIdx.x;
  const int wave = tid >> 5;
  const int lane = tid & 31;
  const int r16  = lane & 15;
  const int hl   = lane >> 4;
  const int bm = blockIdx.x * 128;
  const int bn = blockIdx.y * 256;
  const int m0 = (wave >> 2) * 64;     // 0 or 64
  const int n0 = (wave & 3) * 64;      // 0,64,128,192

  v8f acc[4][4] = {};

  for (int k0 = 0; k0 < DMODEL; k0 += BK) {
    // stage X tile (128 x 32) into LDS as f16 (2 threads per row)
    {
      const int row = tid >> 1;
      const int c   = (tid & 1) * 16;
      if (INF16) {
        const _Float16* X = (const _Float16*)Xv;
        const uint4* gp = (const uint4*)(X + (size_t)(bm + row) * DMODEL + k0 + c);
        uint4 d0 = gp[0], d1 = gp[1];
        *(uint4*)(Xs + row * LDT + c)     = d0;
        *(uint4*)(Xs + row * LDT + c + 8) = d1;
      } else {
        const float* X = (const float*)Xv;
        const float4* gp = (const float4*)(X + (size_t)(bm + row) * DMODEL + k0 + c);
        float4 f0 = gp[0], f1 = gp[1], f2 = gp[2], f3 = gp[3];
        cvt8(Xs + row * LDT + c,     f0, f1);
        cvt8(Xs + row * LDT + c + 8, f2, f3);
      }
      if (k0 + BK < DMODEL) {  // prefetch start of this X row (global_prefetch_b8)
        __builtin_prefetch((const char*)Xv + (size_t)(bm + row) * DMODEL * (INF16 ? 2 : 4), 0, 1);
      }
    }
    // stage W tile (256 rows x 32 cols), f32 -> f16, one row per thread
    {
      const int row = tid;
      const float4* gp = (const float4*)(W + (size_t)(bn + row) * DMODEL + k0);
      float4 f0 = gp[0], f1 = gp[1], f2 = gp[2], f3 = gp[3];
      float4 f4 = gp[4], f5 = gp[5], f6 = gp[6], f7 = gp[7];
      _Float16* dst = Ws + row * LDT;
      cvt8(dst,      f0, f1);
      cvt8(dst + 8,  f2, f3);
      cvt8(dst + 16, f4, f5);
      cvt8(dst + 24, f6, f7);
      if (k0 + BK < DMODEL)
        __builtin_prefetch(W + (size_t)(bn + row) * DMODEL + k0 + BK, 0, 1);
    }
    __syncthreads();

    v16h a[4], b[4];
#pragma unroll
    for (int i = 0; i < 4; ++i) a[i] = load_frag(Xs, LDT, m0 + i * 16, 0);
#pragma unroll
    for (int j = 0; j < 4; ++j) b[j] = load_frag(Ws, LDT, n0 + j * 16, 0);
#pragma unroll
    for (int i = 0; i < 4; ++i)
#pragma unroll
      for (int j = 0; j < 4; ++j) acc[i][j] = wmma16(a[i], b[j], acc[i][j]);
    __syncthreads();
  }

  // store (C layout: VGPR r -> row r / r+8; lane&15 -> column)
#pragma unroll
  for (int i = 0; i < 4; ++i)
#pragma unroll
    for (int j = 0; j < 4; ++j)
#pragma unroll
      for (int r = 0; r < 8; ++r) {
        const int mg = bm + m0 + i * 16 + r + 8 * hl;
        const int ng = bn + n0 + j * 16 + r16;
        const float val = acc[i][j][r] + bias[ng];
        if (MODE == 2) {
          ((float*)Out)[(size_t)mg * DMODEL + ng] = val;
        } else {
          const int bb = mg >> 11, s = mg & (SEQ - 1);
          const int h  = ng >> 6,  d = ng & (DKH - 1);
          _Float16* o = (_Float16*)Out;
          if (MODE == 0)
            o[((size_t)(bb * NH + h) * SEQ + s) * DKH + d] = (_Float16)val;
          else
            o[((size_t)(bb * NH + h) * DKH + d) * SEQ + s] = (_Float16)val;
        }
      }
}

// ---------------------------------------------------------------------------
// Flash attention: grid (B*H, S/64), 128 threads (4 waves).
// 64-key tiles staged in LDS (async DMA path when available).
// Q: [bh][s][dk], K: [bh][s][dk], VT: [bh][dk][s] (all f16)
// Out: [b][s][h*dk+d] f16
// ---------------------------------------------------------------------------
__global__ __launch_bounds__(128) void attn_wmma(const _Float16* __restrict__ Q,
                                                 const _Float16* __restrict__ K,
                                                 const _Float16* __restrict__ VT,
                                                 _Float16* __restrict__ Out) {
  constexpr int BKEY = 64;
  constexpr int LKT = DKH + 8;    // 72 halves
  constexpr int LVT = BKEY + 8;   // 72
  constexpr int LPT = BKEY + 8;   // 72
  __shared__ _Float16 Kt[BKEY * LKT];     // 64 keys x 64 d
  __shared__ _Float16 Vt[DKH * LVT];      // 64 d x 64 keys
  __shared__ _Float16 Pt[4][16 * LPT];    // per-wave P scratch (C->A relayout)

  const int bh   = blockIdx.x;
  const int qb   = blockIdx.y;
  const int tid  = threadIdx.x;
  const int wave = tid >> 5;
  const int lane = tid & 31;
  const int r16  = lane & 15;
  const int hl   = lane >> 4;
  const int q0   = qb * 64 + wave * 16;

  const _Float16* Qb = Q  + (size_t)bh * SEQ * DKH;
  const _Float16* Kb = K  + (size_t)bh * SEQ * DKH;
  const _Float16* Vb = VT + (size_t)bh * DKH * SEQ;

  // resident Q fragments (rows q0..q0+15, d split 0..31 / 32..63)
  const v16h qa0 = load_frag(Qb, DKH, q0, 0);
  const v16h qa1 = load_frag(Qb, DKH, q0, 32);

  v8f O[4] = {};
  float mrun[8], lrun[8];
#pragma unroll
  for (int r = 0; r < 8; ++r) { mrun[r] = -1e30f; lrun[r] = 0.f; }

  const float scale = 0.125f;      // 1/sqrt(dk)
  const int kend = qb * 64 + 64;   // causal upper bound for this query block
  for (int kb = 0; kb < kend; kb += BKEY) {
    // stage K tile (64 keys x 64 d) and V tile (64 d x 64 keys), 2 thr/row
    {
      const int row = tid >> 1;
      const int c   = (tid & 1) * 32;
      const _Float16* gk = Kb + (size_t)(kb + row) * DKH + c;
      const _Float16* gv = Vb + (size_t)row * SEQ + kb + c;
      _Float16* lk = Kt + row * LKT + c;
      _Float16* lv = Vt + row * LVT + c;
#ifdef USE_ASYNC_LDS
#pragma unroll
      for (int u = 0; u < 4; ++u) {
        async_b128(gk + u * 8, lk + u * 8);
        async_b128(gv + u * 8, lv + u * 8);
      }
      wait_async0();
#else
      const uint4* gpk = (const uint4*)gk;
      const uint4* gpv = (const uint4*)gv;
      uint4 k0v = gpk[0], k1v = gpk[1], k2v = gpk[2], k3v = gpk[3];
      uint4 v0v = gpv[0], v1v = gpv[1], v2v = gpv[2], v3v = gpv[3];
      *(uint4*)(lk)      = k0v; *(uint4*)(lk + 8)  = k1v;
      *(uint4*)(lk + 16) = k2v; *(uint4*)(lk + 24) = k3v;
      *(uint4*)(lv)      = v0v; *(uint4*)(lv + 8)  = v1v;
      *(uint4*)(lv + 16) = v2v; *(uint4*)(lv + 24) = v3v;
#endif
    }
    __syncthreads();

    // ---- scores: 16 queries x 64 keys = four 16x16 wmma tiles -------------
    v8f sc[4];
#pragma unroll
    for (int ks = 0; ks < 4; ++ks) {
      v16h b0 = load_frag(Kt, LKT, ks * 16, 0);
      v16h b1 = load_frag(Kt, LKT, ks * 16, 32);
      v8f s = {};
      s = wmma16(qa0, b0, s);
      s = wmma16(qa1, b1, s);
      sc[ks] = s;
    }

    // ---- scale + causal mask + row max (row r+8*hl across 16 lanes) -------
    float tmax[8];
#pragma unroll
    for (int r = 0; r < 8; ++r) {
      const int qrow = q0 + r + 8 * hl;
      float mx = -1e30f;
#pragma unroll
      for (int ks = 0; ks < 4; ++ks) {
        float s = sc[ks][r] * scale;
        if (kb + ks * 16 + r16 > qrow) s = -1e30f;
        sc[ks][r] = s;
        mx = fmaxf(mx, s);
      }
      tmax[r] = mx;
    }
#pragma unroll
    for (int off = 1; off < 16; off <<= 1)
#pragma unroll
      for (int r = 0; r < 8; ++r)
        tmax[r] = fmaxf(tmax[r], __shfl_xor(tmax[r], off, 32));

    // ---- online softmax ---------------------------------------------------
    float alpha[8], psum[8];
#pragma unroll
    for (int r = 0; r < 8; ++r) {
      const float nm = fmaxf(mrun[r], tmax[r]);
      alpha[r] = __expf(mrun[r] - nm);
      mrun[r]  = nm;
      float ps = 0.f;
#pragma unroll
      for (int ks = 0; ks < 4; ++ks) {
        const float p = __expf(sc[ks][r] - nm);
        sc[ks][r] = p;
        ps += p;
      }
      psum[r] = ps;
    }
#pragma unroll
    for (int off = 1; off < 16; off <<= 1)
#pragma unroll
      for (int r = 0; r < 8; ++r)
        psum[r] += __shfl_xor(psum[r], off, 32);
#pragma unroll
    for (int r = 0; r < 8; ++r) lrun[r] = lrun[r] * alpha[r] + psum[r];
#pragma unroll
    for (int j = 0; j < 4; ++j)
#pragma unroll
      for (int r = 0; r < 8; ++r) O[j][r] *= alpha[r];

    // ---- P (C layout) -> A layout via per-wave LDS scratch ---------------
    _Float16* P = Pt[wave];
#pragma unroll
    for (int r = 0; r < 8; ++r)
#pragma unroll
      for (int ks = 0; ks < 4; ++ks)
        P[(r + 8 * hl) * LPT + ks * 16 + r16] = (_Float16)sc[ks][r];
    asm volatile("s_wait_dscnt 0" ::: "memory");  // wave-local LDS RAW fence
    const v16h pa0 = load_frag(P, LPT, 0, 0);
    const v16h pa1 = load_frag(P, LPT, 0, 32);

    // ---- O += P x V -------------------------------------------------------
#pragma unroll
    for (int j = 0; j < 4; ++j) {
      v16h bv0 = load_frag(Vt, LVT, j * 16, 0);
      v16h bv1 = load_frag(Vt, LVT, j * 16, 32);
      O[j] = wmma16(pa0, bv0, O[j]);
      O[j] = wmma16(pa1, bv1, O[j]);
    }
    __syncthreads();
  }

  float inv[8];
#pragma unroll
  for (int r = 0; r < 8; ++r) inv[r] = 1.0f / lrun[r];
  const int bb = bh >> 4, h = bh & (NH - 1);
#pragma unroll
  for (int j = 0; j < 4; ++j)
#pragma unroll
    for (int r = 0; r < 8; ++r) {
      const int qg = q0 + r + 8 * hl;
      const int dg = h * DKH + j * 16 + r16;
      Out[((size_t)(bb * SEQ + qg)) * DMODEL + dg] = (_Float16)(O[j][r] * inv[r]);
    }
}

// ---------------------------------------------------------------------------
extern "C" void kernel_launch(void* const* d_in, const int* in_sizes, int n_in,
                              void* d_out, int out_size, void* d_ws, size_t ws_size,
                              hipStream_t stream) {
  const float* q  = (const float*)d_in[0];
  const float* k  = (const float*)d_in[1];
  const float* v  = (const float*)d_in[2];
  // d_in[3] = causal mask: applied analytically in attn_wmma
  const float* Wq = (const float*)d_in[4];
  const float* bq = (const float*)d_in[5];
  const float* Wk = (const float*)d_in[6];
  const float* bk = (const float*)d_in[7];
  const float* Wv = (const float*)d_in[8];
  const float* bv = (const float*)d_in[9];
  const float* Wo = (const float*)d_in[10];
  const float* bo = (const float*)d_in[11];
  float* out = (float*)d_out;

  char* ws = (char*)d_ws;
  const size_t QH = (size_t)MROWS * DMODEL * sizeof(_Float16);  // 8 MB each
  _Float16* qw = (_Float16*)(ws);            // Q  [bh][s][dk]
  _Float16* kw = (_Float16*)(ws + QH);       // K  [bh][s][dk]
  _Float16* vw = (_Float16*)(ws + 2 * QH);   // V^T[bh][dk][s]
  _Float16* aw = (_Float16*)(ws + 3 * QH);   // attn out [b][s][D]

  const dim3 gg(MROWS / 128, DMODEL / 256);  // 32 x 4 blocks, 256 threads
  gemm_wmma<0, false><<<gg, 256, 0, stream>>>(q, Wq, bq, qw);
  gemm_wmma<0, false><<<gg, 256, 0, stream>>>(k, Wk, bk, kw);
  gemm_wmma<1, false><<<gg, 256, 0, stream>>>(v, Wv, bv, vw);

  attn_wmma<<<dim3(2 * NH, SEQ / 64), 128, 0, stream>>>(qw, kw, vw, aw);

  gemm_wmma<2, true><<<gg, 256, 0, stream>>>(aw, Wo, bo, out);
}
